// NeuralGraphODEController_53850299957782
// MI455X (gfx1250) — matrix operand order repaired and checked by reading, not compiled, not run on hardware
//
#include <hip/hip_runtime.h>
#include <hip/hip_bf16.h>
#include <math.h>

typedef __bf16 bf16_t;
typedef __attribute__((ext_vector_type(16))) __bf16 v16bf;
typedef __attribute__((ext_vector_type(8)))  __bf16 v8bf;
typedef __attribute__((ext_vector_type(8)))  float  v8f;

#define NODES   64
#define HIDN    64
#define TSTEPS  128
#define BATCH   32
#define OUTC    6
#define THREADS 512

#if __has_builtin(__builtin_amdgcn_tanhf)
#define TANHF(x) __builtin_amdgcn_tanhf(x)
#elif __has_builtin(__builtin_amdgcn_tanh_f32)
#define TANHF(x) __builtin_amdgcn_tanh_f32(x)
#else
#define TANHF(x) tanhf(x)
#endif

__device__ __forceinline__ v16bf cat16(v8bf lo, v8bf hi) {
    return __builtin_shufflevector(lo, hi, 0, 1, 2, 3, 4, 5, 6, 7,
                                   8, 9, 10, 11, 12, 13, 14, 15);
}

// A-fragment (16x32 bf16, ISA 7.12.2): lane -> m=L%16, g=L/16.
// Elements 0..15 = K[16g .. 16g+15] (matches swizzled B)
// -> one contiguous 32B run => 2x ds_load_b128.
__device__ __forceinline__ v16bf load_a_frag(const bf16_t* A, int lda, int r0, int c0, int lane) {
    const int m = lane & 15, g = lane >> 4;
    const bf16_t* base = A + (r0 + m) * lda + c0 + 16 * g;
    v8bf lo = *(const v8bf*)(base);
    v8bf hi = *(const v8bf*)(base + 8);
    return cat16(lo, hi);
}

// Swizzled B storage: frag block = ((k0/32)*4 + ntile); within block, lane*16
// contiguous elements j = B[k0 + 16*(lane/16) + j][nt*16 + lane%16].
__device__ __forceinline__ v16bf load_b_sw(const bf16_t* Bsw, int k0, int nt, int lane) {
    const bf16_t* p = Bsw + ((((k0 >> 5) << 2) | nt) * 32 + lane) * 16;
    v8bf lo = *(const v8bf*)(p);
    v8bf hi = *(const v8bf*)(p + 8);
    return cat16(lo, hi);
}

// row-major (row,col) -> swizzled-B index
__device__ __forceinline__ int bsw_idx(int row, int col) {
    return ((((row >> 5) << 2) | (col >> 4)) * 32 + ((row >> 4) & 1) * 16 + (col & 15)) * 16
           + (row & 15);
}

__device__ __forceinline__ v8f wmma_acc(v16bf a, v16bf b, v8f acc) {
    return __builtin_amdgcn_wmma_f32_16x16x32_bf16(false, a, false, b, (short)0, acc,
                                                   false, false);
}

// C += A(MxK row-major) * B(KxN swizzled), one 16x16 tile per wave
__device__ __forceinline__ v8f gemm_sw(const bf16_t* A, int lda, const bf16_t* Bsw,
                                       int K, int mt, int nt, v8f acc, int lane) {
    for (int k0 = 0; k0 < K; k0 += 32) {
        v16bf a = load_a_frag(A, lda, mt * 16, k0, lane);
        v16bf b = load_b_sw(Bsw, k0, nt, lane);
        acc = wmma_acc(a, b, acc);
    }
    return acc;
}

__global__ void __launch_bounds__(THREADS)
NeuralGraphODEController_53850299957782_kernel(
    const float* __restrict__ frames, const float* __restrict__ adj,
    const float* __restrict__ hs,
    const float* __restrict__ encW_g, const float* __restrict__ encb_g,
    const float* __restrict__ Wh_g, const float* __restrict__ Wu_g,
    const float* __restrict__ odeb_g,
    const float* __restrict__ d1W, const float* __restrict__ d1b,
    const float* __restrict__ d2W, const float* __restrict__ d2b,
    const float* __restrict__ d3W, const float* __restrict__ d3b,
    const float* __restrict__ oscale, const float* __restrict__ obias,
    float* __restrict__ out)
{
    extern __shared__ char smem_raw[];
    float* hrow = (float*)smem_raw;        // 4096 (post-step h, for decoder pool)
    float* odeb = hrow + 4096;             // 64
    float* encb = odeb + 64;               // 64
    float* mvec = encb + 64;               // 64
    float* a1   = mvec + 64;               // 128
    float* a2   = a1 + 128;                // 64   (fp32 total: 4480 -> 17920 B)
    bf16_t* Whsw   = (bf16_t*)(a2 + 64);   // 3*4096 swizzled
    bf16_t* Wusw   = Whsw + 3 * 4096;      // 3*4096 swizzled
    bf16_t* encWsw = Wusw + 3 * 4096;      // 8192 (128x64) swizzled
    bf16_t* S1row  = encWsw + 8192;        // 4096 each below
    bf16_t* S1sw   = S1row + 4096;
    bf16_t* S2row  = S1sw + 4096;
    bf16_t* xtrow  = S2row + 4096;
    bf16_t* xtsw   = xtrow + 4096;
    bf16_t* Y1row  = xtsw + 4096;
    bf16_t* Y2row  = Y1row + 4096;
    bf16_t* hbrow  = Y2row + 4096;
    bf16_t* hbsw   = hbrow + 4096;
    bf16_t* Xb     = Y1row;                // overlay: 8192 (64x128) over Y1row+Y2row

    const int tid  = threadIdx.x;
    const int lane = tid & 31;
    const int wave = tid >> 5;
    const int mt = wave >> 2, nt = wave & 3;
    const int b  = blockIdx.x;

    // This thread's 8 C-layout elements: row = rowb + v (rowb % 8 == 0), fixed col
    const int rowb = mt * 16 + 8 * (lane >> 4);
    const int col  = nt * 16 + (lane & 15);

    // staging mapping: 8-row run of one column per thread (coalesced global reads)
    const int sc = tid & 63;            // column
    const int sq = (tid >> 6) * 8;      // first row of run

    // ---- one-time setup ----
    for (int e = tid; e < 3 * 4096; e += THREADS) {
        const int mat = e >> 12, r = (e >> 6) & 63, c = e & 63;
        const int dst = mat * 4096 + bsw_idx(r, c);
        Whsw[dst] = (bf16_t)Wh_g[e];
        Wusw[dst] = (bf16_t)Wu_g[e];
    }
    for (int e = tid; e < 8192; e += THREADS) {
        const int r = e >> 6, c = e & 63;
        encWsw[bsw_idx(r, c)] = (bf16_t)encW_g[e];
    }
    if (tid < 64) { odeb[tid] = odeb_g[tid]; encb[tid] = encb_g[tid]; }

    // ODE state in registers (h, hnew, hh, U), one element set per thread
    float hreg[8], hnreg[8], hhreg[8], ureg[8];
    {
        v8bf pk;
#pragma unroll
        for (int v = 0; v < 8; ++v) {
            const float h0 = hs[(size_t)b * 4096 + (rowb + v) * 64 + col];
            hreg[v] = h0; hhreg[v] = h0;
            pk[v] = (bf16_t)h0;
        }
        *(v8bf*)(hbsw + bsw_idx(rowb, col)) = pk;
#pragma unroll
        for (int v = 0; v < 8; ++v) hbrow[(rowb + v) * 64 + col] = pk[v];
    }
    __syncthreads();

    // hoist Wh B-fragments into registers (constant for whole kernel)
    v16bf whf[3][2];
#pragma unroll
    for (int m = 0; m < 3; ++m)
#pragma unroll
        for (int kt = 0; kt < 2; ++kt)
            whf[m][kt] = load_b_sw(Whsw + m * 4096, kt * 32, nt, lane);

    const float dt = 1.0f / 4.0f;   // N_ODE_STEPS = 4
    const float obia = odeb[col];

#pragma unroll 1
    for (int t = 0; t < TSTEPS; ++t) {
        const float* fr = frames + (size_t)(b * TSTEPS + t) * 8192;
        const float* At = adj    + (size_t)(b * TSTEPS + t) * 4096;

        if (t + 1 < TSTEPS) {                       // one cacheline per lane prefetch
            const int e = tid * 16;
            if (e < 8192) __builtin_prefetch(fr + 8192 + e, 0, 1);
            if (e < 4096) __builtin_prefetch(At + 4096 + e, 0, 1);
        }

        // stage X (frames slice transposed, 64x128 row-major)
        for (int e = tid; e < 8192; e += THREADS) {
            const int c = e >> 6, n = e & 63;       // mem [c][n] -> X[n][c]
            Xb[n * 128 + c] = (bf16_t)fr[e];
        }
        // stage S1 (row + one vector-swizzled store per thread)
        {
            v8bf pk;
#pragma unroll
            for (int v = 0; v < 8; ++v) {
                pk[v] = (bf16_t)At[(sq + v) * 64 + sc];
                S1row[(sq + v) * 64 + sc] = pk[v];
            }
            *(v8bf*)(S1sw + bsw_idx(sq, sc)) = pk;
        }
        __syncthreads();

        // S1 A-fragments: invariant for the whole timestep -> registers
        v16bf s1a[2];
#pragma unroll
        for (int kt = 0; kt < 2; ++kt)
            s1a[kt] = load_a_frag(S1row, 64, mt * 16, kt * 32, lane);

        // encoder: xt = X @ encW + encb  (dual-store: A-use and B-use layouts)
        {
            v8f acc = {};
            acc = gemm_sw(Xb, 128, encWsw, 128, mt, nt, acc, lane);
            v8bf pk;
#pragma unroll
            for (int v = 0; v < 8; ++v) {
                pk[v] = (bf16_t)(acc[v] + encb[col]);
                xtrow[(rowb + v) * 64 + col] = pk[v];
            }
            *(v8bf*)(xtsw + bsw_idx(rowb, col)) = pk;
        }
        __syncthreads();

        // S2 = S1 @ S1  (A frags preloaded)
        {
            v8f acc = {};
#pragma unroll
            for (int kt = 0; kt < 2; ++kt)
                acc = wmma_acc(s1a[kt], load_b_sw(S1sw, kt * 32, nt, lane), acc);
#pragma unroll
            for (int v = 0; v < 8; ++v)
                S2row[(rowb + v) * 64 + col] = (bf16_t)acc[v];
        }
        __syncthreads();

        // S2 A-fragments: invariant for the whole timestep -> registers
        v16bf s2a[2];
#pragma unroll
        for (int kt = 0; kt < 2; ++kt)
            s2a[kt] = load_a_frag(S2row, 64, mt * 16, kt * 32, lane);

        // Y1 = S1 @ xt ; Y2 = S2 @ xt  (shared B fragment)
        {
            v8f acc1 = {}, acc2 = {};
#pragma unroll
            for (int kt = 0; kt < 2; ++kt) {
                v16bf bx = load_b_sw(xtsw, kt * 32, nt, lane);
                acc1 = wmma_acc(s1a[kt], bx, acc1);
                acc2 = wmma_acc(s2a[kt], bx, acc2);
            }
#pragma unroll
            for (int v = 0; v < 8; ++v) {
                const int e = (rowb + v) * 64 + col;
                Y1row[e] = (bf16_t)acc1[v];
                Y2row[e] = (bf16_t)acc2[v];
            }
        }
        __syncthreads();

        // U = xt@Wu0 + Y1@Wu1 + Y2@Wu2 + ode_b  -> registers
        {
            v8f acc = {};
            acc = gemm_sw(xtrow, 64, Wusw,            64, mt, nt, acc, lane);
            acc = gemm_sw(Y1row, 64, Wusw + 4096,     64, mt, nt, acc, lane);
            acc = gemm_sw(Y2row, 64, Wusw + 2 * 4096, 64, mt, nt, acc, lane);
#pragma unroll
            for (int v = 0; v < 8; ++v) ureg[v] = acc[v] + obia;
        }
        __syncthreads();

        // ---- RK4: 4 steps x 4 evals; f(hh) = -hh + tanh(gconv(hh) + U) ----
#pragma unroll 1
        for (int s = 0; s < 4; ++s) {
#pragma unroll 1
            for (int ev = 0; ev < 4; ++ev) {
                // T1 = S1@hh, T2 = S2@hh  (A frags in regs, shared B fragment)
                v8f t1 = {}, t2 = {};
#pragma unroll
                for (int kt = 0; kt < 2; ++kt) {
                    v16bf bh = load_b_sw(hbsw, kt * 32, nt, lane);
                    t1 = wmma_acc(s1a[kt], bh, t1);
                    t2 = wmma_acc(s2a[kt], bh, t2);
                }
#pragma unroll
                for (int v = 0; v < 8; ++v) {
                    const int e = (rowb + v) * 64 + col;
                    Y1row[e] = (bf16_t)t1[v];
                    Y2row[e] = (bf16_t)t2[v];
                }
                __syncthreads();

                // Z = hh@Wh0 + T1@Wh1 + T2@Wh2  (B frags in registers)
                v8f z = {};
#pragma unroll
                for (int kt = 0; kt < 2; ++kt)
                    z = wmma_acc(load_a_frag(hbrow, 64, mt * 16, kt * 32, lane),
                                 whf[0][kt], z);
#pragma unroll
                for (int kt = 0; kt < 2; ++kt)
                    z = wmma_acc(load_a_frag(Y1row, 64, mt * 16, kt * 32, lane),
                                 whf[1][kt], z);
#pragma unroll
                for (int kt = 0; kt < 2; ++kt)
                    z = wmma_acc(load_a_frag(Y2row, 64, mt * 16, kt * 32, lane),
                                 whf[2][kt], z);

                const float wcoef = (ev == 0 || ev == 3) ? dt / 6.0f : dt / 3.0f;
                const float ccoef = (ev == 2) ? dt : 0.5f * dt;
                v8bf pk;
#pragma unroll
                for (int v = 0; v < 8; ++v) {
                    const float k  = TANHF(z[v] + ureg[v]) - hhreg[v];
                    const float hn = ((ev == 0) ? hreg[v] : hnreg[v]) + wcoef * k;
                    hnreg[v] = hn;
                    float nh;
                    if (ev < 3) {
                        nh = hreg[v] + ccoef * k;      // next stage point
                    } else {
                        nh = hn;                       // step complete
                        hreg[v] = hn;
                        if (s == 3) hrow[(rowb + v) * 64 + col] = hn;  // decoder pool
                    }
                    hhreg[v] = nh;
                    pk[v] = (bf16_t)nh;
                }
                *(v8bf*)(hbsw + bsw_idx(rowb, col)) = pk;
#pragma unroll
                for (int v = 0; v < 8; ++v)
                    hbrow[(rowb + v) * 64 + col] = pk[v];
                __syncthreads();
            }
        }

        // ---- decoder MLP on mean-pooled hidden ----
        if (tid < 64) {
            float sv = 0.f;
            for (int n = 0; n < 64; ++n) sv += hrow[n * 64 + tid];
            mvec[tid] = sv * (1.0f / 64.0f);
        }
        __syncthreads();
        if (tid < 128) {
            float sv = d1b[tid];
            for (int d = 0; d < 64; ++d) sv += mvec[d] * d1W[d * 128 + tid];
            a1[tid] = fmaxf(sv, 0.f);
        }
        __syncthreads();
        if (tid < 64) {
            float sv = d2b[tid];
            for (int i = 0; i < 128; ++i) sv += a1[i] * d2W[i * 64 + tid];
            a2[tid] = fmaxf(sv, 0.f);
        }
        __syncthreads();
        if (tid < OUTC) {
            float sv = d3b[tid];
            for (int i = 0; i < 64; ++i) sv += a2[i] * d3W[i * OUTC + tid];
            out[(size_t)(b * TSTEPS + t) * OUTC + tid] = sv * oscale[tid] + obias[tid];
        }
        __syncthreads();
    }

    // final_hidden (B,1,N,HID) appended after controls (B,T,OUT)
#pragma unroll
    for (int v = 0; v < 8; ++v)
        out[(size_t)BATCH * TSTEPS * OUTC + (size_t)b * 4096 + (rowb + v) * 64 + col] = hreg[v];
}

extern "C" void kernel_launch(void* const* d_in, const int* in_sizes, int n_in,
                              void* d_out, int out_size, void* d_ws, size_t ws_size,
                              hipStream_t stream) {
    (void)in_sizes; (void)n_in; (void)out_size; (void)d_ws; (void)ws_size;
    const float* frames = (const float*)d_in[0];
    const float* adjac  = (const float*)d_in[1];
    const float* hidden = (const float*)d_in[2];
    const float* encW   = (const float*)d_in[3];
    const float* encb   = (const float*)d_in[4];
    const float* Wh     = (const float*)d_in[5];
    const float* Wu     = (const float*)d_in[6];
    const float* odeb   = (const float*)d_in[7];
    const float* d1W    = (const float*)d_in[8];
    const float* d1b    = (const float*)d_in[9];
    const float* d2W    = (const float*)d_in[10];
    const float* d2b    = (const float*)d_in[11];
    const float* d3W    = (const float*)d_in[12];
    const float* d3b    = (const float*)d_in[13];
    const float* oscale = (const float*)d_in[14];
    const float* obias  = (const float*)d_in[15];
    float* out = (float*)d_out;

    // LDS: fp32 17,920 B + bf16 139,264 B = 157,184 B (< 320 KB/WGP)
    const size_t shmem = 157184;
    NeuralGraphODEController_53850299957782_kernel<<<BATCH, THREADS, shmem, stream>>>(
        frames, adjac, hidden, encW, encb, Wh, Wu, odeb,
        d1W, d1b, d2W, d2b, d3W, d3b, oscale, obias, out);
}
